// SFGAT_SE_LONG_16939351015639
// MI455X (gfx1250) — matrix operand order repaired and checked by reading, not compiled, not run on hardware
//
#include <hip/hip_runtime.h>
#include <math.h>

typedef __bf16 bf16_t;
typedef __attribute__((ext_vector_type(16))) __bf16 v16bf;
typedef __attribute__((ext_vector_type(8)))  float  v8f;

#define N_FEAT 445

__device__ __forceinline__ float fast_rcp(float x) {
#if __has_builtin(__builtin_amdgcn_rcpf)
  return __builtin_amdgcn_rcpf(x);   // v_rcp_f32, no Newton refinement chain
#else
  return 1.f / x;
#endif
}

__device__ __forceinline__ float sigmoidf_(float x) {
  return fast_rcp(1.f + __expf(-x)); // v_exp_f32 + v_rcp_f32
}

__device__ __forceinline__ float fast_tanh(float x) {
#if __has_builtin(__builtin_amdgcn_tanhf)
  return __builtin_amdgcn_tanhf(x);  // gfx1250 hardware v_tanh_f32
#else
  // tanh(x) = 1 - 2/(exp(2x)+1)
  return __builtin_fmaf(-2.f, fast_rcp(__expf(2.f * x) + 1.f), 1.f);
#endif
}

__device__ __forceinline__ float lrelu(float x, float s) { return x > 0.f ? x : s * x; }

__device__ __forceinline__ v8f wmma_bf16(v16bf a, v16bf b, v8f c) {
  return __builtin_amdgcn_wmma_f32_16x16x32_bf16(false, a, false, b, (short)0, c, false, false);
}

// A-fragment (16x32 bf16) from an LDS tile (16 rows x ldk), k-chunk kc.
// Per ISA: lanes 0-15 hold K = kc+[0..7] and kc+[16..23]; lanes 16-31 hold K+8.
__device__ __forceinline__ v16bf frag_a_lds(const bf16_t* base, int ldk, int kc) {
  const int lane = threadIdx.x & 31;
  const int half = lane >> 4;
  const int m    = lane & 15;
  const bf16_t* p0 = base + m * ldk + kc + half * 8;
  v16bf a;
#pragma unroll
  for (int e = 0; e < 8; ++e) a[e] = p0[e];
#pragma unroll
  for (int e = 0; e < 8; ++e) a[8 + e] = p0[16 + e];
  return a;
}

// B-fragment (32x16 bf16) for out = x @ W^T, W row-major [outs][ldk] in global (bf16).
// Per ISA: lanes 0-15 hold K = kc+[0..7] and kc+[16..23]; lanes 16-31 hold K+8; N = lane&15.
__device__ __forceinline__ v16bf frag_b_glb(const bf16_t* __restrict__ W, int ldk, int nbase, int kc) {
  const int lane = threadIdx.x & 31;
  const int half = lane >> 4;
  const int nn   = lane & 15;
  const bf16_t* p = W + (size_t)(nbase + nn) * ldk + kc + half * 8;
  v16bf b;
#pragma unroll
  for (int e = 0; e < 8; ++e) { b[e] = p[e]; b[8 + e] = p[16 + e]; }
  return b;
}

// ---------------- weight conversion (fp32 -> padded bf16) ----------------
__global__ void k_cvt(const float* __restrict__ src, bf16_t* __restrict__ dst,
                      int rows, int cols, int ldk, int rowsPad) {
  int i = blockIdx.x * blockDim.x + threadIdx.x;
  int total = rowsPad * ldk;
  if (i >= total) return;
  int r = i / ldk, k = i - r * ldk;
  float v = (r < rows && k < cols) ? src[(size_t)r * cols + k] : 0.f;
  dst[i] = (bf16_t)v;
}

__global__ void k_bsum(const float* __restrict__ a, const float* __restrict__ b,
                       float* __restrict__ o, int nel) {
  int i = blockIdx.x * blockDim.x + threadIdx.x;
  if (i < nel) o[i] = a[i] + b[i];
}

// ---------------- sec1 + sec2 fused (40->128 sigmoid, 128->128 sigmoid) ----------------
__global__ __launch_bounds__(256) void k_sec(const float* __restrict__ xall,
    const bf16_t* __restrict__ w1, const float* __restrict__ b1,
    const bf16_t* __restrict__ w2, const float* __restrict__ b2,
    float* __restrict__ x0) {
  __shared__ bf16_t sx[128 * 64];
  __shared__ bf16_t smid[128 * 128];
  const int nb = blockIdx.x * 128;
  for (int i = threadIdx.x; i < 128 * 64; i += 256) {
    int r = i >> 6, k = i & 63;
    float v = (k < 40) ? xall[(size_t)(nb + r) * N_FEAT + 16 + k] : 0.f;
    sx[i] = (bf16_t)v;
  }
  __syncthreads();
  const int rbase = (threadIdx.x >> 5) * 16;
  const int lane = threadIdx.x & 31, half = lane >> 4, col = lane & 15;
  for (int nt = 0; nt < 8; ++nt) {
    v8f acc = {};
    for (int kc = 0; kc < 64; kc += 32)
      acc = wmma_bf16(frag_a_lds(sx + rbase * 64, 64, kc), frag_b_glb(w1, 64, nt * 16, kc), acc);
    float bias = b1[nt * 16 + col];
#pragma unroll
    for (int r = 0; r < 8; ++r) {
      int m = rbase + r + half * 8;
      smid[m * 128 + nt * 16 + col] = (bf16_t)sigmoidf_(acc[r] + bias);
    }
  }
  __syncthreads();
  for (int nt = 0; nt < 8; ++nt) {
    v8f acc = {};
    for (int kc = 0; kc < 128; kc += 32)
      acc = wmma_bf16(frag_a_lds(smid + rbase * 128, 128, kc), frag_b_glb(w2, 128, nt * 16, kc), acc);
    float bias = b2[nt * 16 + col];
#pragma unroll
    for (int r = 0; r < 8; ++r) {
      int m = rbase + r + half * 8;
      x0[(size_t)(nb + m) * 128 + nt * 16 + col] = sigmoidf_(acc[r] + bias);
    }
  }
}

// ---------------- GAT h = x @ W^T  (+ es/ed dot products, init max/sum, zero accum) ----------------
template <int OUT>
__global__ __launch_bounds__(256) void k_gat_h(const float* __restrict__ xin,
    const float* __restrict__ in_bias,   // nullable: apply lrelu(x+bias,0.01) at staging
    const bf16_t* __restrict__ W, const float* __restrict__ asrc, const float* __restrict__ adst,
    float* __restrict__ h, float* __restrict__ es, float* __restrict__ ed,
    float* __restrict__ mmax, float* __restrict__ ssum, float* __restrict__ accum) {
  __shared__ bf16_t sx[128 * 128];
  __shared__ float sav[2 * OUT];
  const int nb = blockIdx.x * 128;
  for (int i = threadIdx.x; i < OUT; i += 256) { sav[i] = asrc[i]; sav[OUT + i] = adst[i]; }
  for (int i = threadIdx.x; i < 128 * 128; i += 256) {
    int r = i >> 7, k = i & 127;
    float v = xin[(size_t)(nb + r) * 128 + k];
    if (in_bias) v = lrelu(v + in_bias[k], 0.01f);
    sx[i] = (bf16_t)v;
  }
  for (int i = threadIdx.x; i < 128 * OUT; i += 256) accum[(size_t)nb * OUT + i] = 0.f;
  __syncthreads();
  const int rbase = (threadIdx.x >> 5) * 16;
  const int lane = threadIdx.x & 31, half = lane >> 4, col = lane & 15;
  float pes[8], ped[8];
#pragma unroll
  for (int r = 0; r < 8; ++r) { pes[r] = 0.f; ped[r] = 0.f; }
  for (int nt = 0; nt < OUT / 16; ++nt) {
    v8f acc = {};
    for (int kc = 0; kc < 128; kc += 32)
      acc = wmma_bf16(frag_a_lds(sx + rbase * 128, 128, kc), frag_b_glb(W, 128, nt * 16, kc), acc);
    float va = sav[nt * 16 + col], vd = sav[OUT + nt * 16 + col];
#pragma unroll
    for (int r = 0; r < 8; ++r) {
      int m = rbase + r + half * 8;
      float v = acc[r];
      h[(size_t)(nb + m) * OUT + nt * 16 + col] = v;
      pes[r] += v * va;
      ped[r] += v * vd;
    }
  }
#pragma unroll
  for (int r = 0; r < 8; ++r) {
    float e1 = pes[r], e2 = ped[r];
    for (int off = 8; off > 0; off >>= 1) {
      e1 += __shfl_xor(e1, off, 16);
      e2 += __shfl_xor(e2, off, 16);
    }
    if (col == 0) {
      int node = nb + rbase + r + half * 8;
      es[node] = e1; ed[node] = e2;
      mmax[node] = -__builtin_huge_valf();
      ssum[node] = 0.f;
    }
  }
}

// ---------------- edge passes: softmax over incoming edges of dst ----------------
__device__ __forceinline__ void edge_sd(int e, int nE, const int* __restrict__ esrc,
                                        const int* __restrict__ edst, int& s, int& d) {
  s = (e < nE) ? esrc[e] : (e - nE);
  d = (e < nE) ? edst[e] : (e - nE);
}

__global__ void k_edge_max(const int* __restrict__ esrc, const int* __restrict__ edst,
                           const float* __restrict__ es, const float* __restrict__ ed,
                           float* __restrict__ mmax, int nE, int nTot) {
  int e = blockIdx.x * 256 + threadIdx.x;
  if (e >= nTot) return;
  int s, d; edge_sd(e, nE, esrc, edst, s, d);
  float c = lrelu(es[s] + ed[d], 0.2f);
  if (c >= 0.f) atomicMax((int*)(mmax + d), __float_as_int(c));
  else          atomicMin((unsigned int*)(mmax + d), __float_as_uint(c));
}

__global__ void k_edge_sum(const int* __restrict__ esrc, const int* __restrict__ edst,
                           const float* __restrict__ es, const float* __restrict__ ed,
                           const float* __restrict__ mmax, float* __restrict__ ssum,
                           int nE, int nTot) {
  int e = blockIdx.x * 256 + threadIdx.x;
  if (e >= nTot) return;
  int s, d; edge_sd(e, nE, esrc, edst, s, d);
  float c = lrelu(es[s] + ed[d], 0.2f);
  atomicAdd(ssum + d, __expf(c - mmax[d]));
}

template <int OUT>
__global__ __launch_bounds__(256) void k_edge_scatter(const int* __restrict__ esrc,
    const int* __restrict__ edst, const float* __restrict__ es, const float* __restrict__ ed,
    const float* __restrict__ mmax, const float* __restrict__ ssum,
    const float* __restrict__ h, float* __restrict__ accum, int nE, int nTot) {
  int e = blockIdx.x * 8 + (threadIdx.x >> 5);
  if (e >= nTot) return;
  const int lane = threadIdx.x & 31;
  int s, d; edge_sd(e, nE, esrc, edst, s, d);
  float c = lrelu(es[s] + ed[d], 0.2f);
  float alpha = __expf(c - mmax[d]) * fast_rcp(ssum[d] + 1e-16f);
  const float4* hp = (const float4*)(h + (size_t)s * OUT);
  float* ap = accum + (size_t)d * OUT;
  for (int j = lane; j < OUT / 4; j += 32) {
    float4 hv = hp[j];
    atomicAdd(ap + 4 * j + 0, alpha * hv.x);
    atomicAdd(ap + 4 * j + 1, alpha * hv.y);
    atomicAdd(ap + 4 * j + 2, alpha * hv.z);
    atomicAdd(ap + 4 * j + 3, alpha * hv.w);
  }
}

// ---------------- gated (pseudo-LSTM) layer: h = sig(o)*tanh(sig(i)*tanh(g)); f unused ----------------
__device__ __forceinline__ void gated_layer(const bf16_t* sA, int ldk,
    const bf16_t* __restrict__ W, const float* __restrict__ bs, bf16_t* sOut,
    int rbase, int half, int col) {
  for (int t = 0; t < 4; ++t) {
    v8f ai = {}, ag = {}, ao = {};
    for (int kc = 0; kc < ldk; kc += 32) {
      v16bf a = frag_a_lds(sA + rbase * ldk, ldk, kc);
      ai = wmma_bf16(a, frag_b_glb(W, ldk, t * 16, kc), ai);
      ag = wmma_bf16(a, frag_b_glb(W, ldk, 128 + t * 16, kc), ag);
      ao = wmma_bf16(a, frag_b_glb(W, ldk, 192 + t * 16, kc), ao);
    }
    float bi = bs[t * 16 + col], bg = bs[128 + t * 16 + col], bo = bs[192 + t * 16 + col];
#pragma unroll
    for (int r = 0; r < 8; ++r) {
      int m = rbase + r + half * 8;
      float cc = sigmoidf_(ai[r] + bi) * fast_tanh(ag[r] + bg);
      float hv = sigmoidf_(ao[r] + bo) * fast_tanh(cc);
      sOut[m * 64 + t * 16 + col] = (bf16_t)hv;
    }
  }
}

__global__ __launch_bounds__(256) void k_lstm(const float* __restrict__ xall,
    const bf16_t* __restrict__ w0, const float* __restrict__ bs0,
    const bf16_t* __restrict__ w1, const float* __restrict__ bs1,
    const bf16_t* __restrict__ w2, const float* __restrict__ bs2,
    const bf16_t* __restrict__ wt1, const float* __restrict__ bt1,
    const bf16_t* __restrict__ wt2, const float* __restrict__ bt2,
    float* __restrict__ tout) {
  __shared__ bf16_t sin0[128 * 32];
  __shared__ bf16_t shA[128 * 64];
  __shared__ bf16_t shB[128 * 64];
  const int nb = blockIdx.x * 128;
  for (int i = threadIdx.x; i < 128 * 32; i += 256) {
    int r = i >> 5, k = i & 31;
    float v = (k < 24) ? xall[(size_t)(nb + r) * N_FEAT + 421 + k] : 0.f;
    sin0[i] = (bf16_t)v;
  }
  __syncthreads();
  const int rbase = (threadIdx.x >> 5) * 16;
  const int lane = threadIdx.x & 31, half = lane >> 4, col = lane & 15;
  gated_layer(sin0, 32, w0, bs0, shA, rbase, half, col);
  gated_layer(shA, 64, w1, bs1, shB, rbase, half, col);
  gated_layer(shB, 64, w2, bs2, shA, rbase, half, col);
  // time1 (64->64, lrelu 0.01) -> shB
  for (int t = 0; t < 4; ++t) {
    v8f acc = {};
    for (int kc = 0; kc < 64; kc += 32)
      acc = wmma_bf16(frag_a_lds(shA + rbase * 64, 64, kc), frag_b_glb(wt1, 64, t * 16, kc), acc);
    float b = bt1[t * 16 + col];
#pragma unroll
    for (int r = 0; r < 8; ++r) {
      int m = rbase + r + half * 8;
      shB[m * 64 + t * 16 + col] = (bf16_t)lrelu(acc[r] + b, 0.01f);
    }
  }
  // time2 (64->64, lrelu 0.01) -> tout (global fp32)
  for (int t = 0; t < 4; ++t) {
    v8f acc = {};
    for (int kc = 0; kc < 64; kc += 32)
      acc = wmma_bf16(frag_a_lds(shB + rbase * 64, 64, kc), frag_b_glb(wt2, 64, t * 16, kc), acc);
    float b = bt2[t * 16 + col];
#pragma unroll
    for (int r = 0; r < 8; ++r) {
      int m = rbase + r + half * 8;
      tout[(size_t)(nb + m) * 64 + t * 16 + col] = lrelu(acc[r] + b, 0.01f);
    }
  }
}

// ---------------- final: concat(gat3, t) -> lin1 relu -> lin2 relu -> lin3 ----------------
__global__ __launch_bounds__(256) void k_final(const float* __restrict__ accum3,
    const float* __restrict__ gb3, const float* __restrict__ tbuf,
    const bf16_t* __restrict__ wl1, const float* __restrict__ bl1,
    const bf16_t* __restrict__ wl2, const float* __restrict__ bl2,
    const bf16_t* __restrict__ wl3, const float* __restrict__ bl3,
    float* __restrict__ out) {
  __shared__ bf16_t sx[128 * 128];
  __shared__ bf16_t sm1[128 * 64];
  __shared__ bf16_t sm2[128 * 64];
  const int nb = blockIdx.x * 128;
  for (int i = threadIdx.x; i < 128 * 64; i += 256) {
    int r = i >> 6, k = i & 63;
    sx[r * 128 + k]      = (bf16_t)lrelu(accum3[(size_t)(nb + r) * 64 + k] + gb3[k], 0.01f);
    sx[r * 128 + 64 + k] = (bf16_t)tbuf[(size_t)(nb + r) * 64 + k];
  }
  __syncthreads();
  const int rbase = (threadIdx.x >> 5) * 16;
  const int lane = threadIdx.x & 31, half = lane >> 4, col = lane & 15;
  for (int t = 0; t < 4; ++t) {
    v8f acc = {};
    for (int kc = 0; kc < 128; kc += 32)
      acc = wmma_bf16(frag_a_lds(sx + rbase * 128, 128, kc), frag_b_glb(wl1, 128, t * 16, kc), acc);
    float b = bl1[t * 16 + col];
#pragma unroll
    for (int r = 0; r < 8; ++r) {
      int m = rbase + r + half * 8;
      sm1[m * 64 + t * 16 + col] = (bf16_t)fmaxf(acc[r] + b, 0.f);
    }
  }
  for (int t = 0; t < 4; ++t) {
    v8f acc = {};
    for (int kc = 0; kc < 64; kc += 32)
      acc = wmma_bf16(frag_a_lds(sm1 + rbase * 64, 64, kc), frag_b_glb(wl2, 64, t * 16, kc), acc);
    float b = bl2[t * 16 + col];
#pragma unroll
    for (int r = 0; r < 8; ++r) {
      int m = rbase + r + half * 8;
      sm2[m * 64 + t * 16 + col] = (bf16_t)fmaxf(acc[r] + b, 0.f);
    }
  }
  {
    v8f acc = {};
    for (int kc = 0; kc < 64; kc += 32)
      acc = wmma_bf16(frag_a_lds(sm2 + rbase * 64, 64, kc), frag_b_glb(wl3, 64, 0, kc), acc);
    if (col < 4) {
      float b = bl3[col];
#pragma unroll
      for (int r = 0; r < 8; ++r) {
        int m = rbase + r + half * 8;
        out[(size_t)(nb + m) * 4 + col] = acc[r] + b;
      }
    }
  }
}

extern "C" void kernel_launch(void* const* d_in, const int* in_sizes, int n_in,
                              void* d_out, int out_size, void* d_ws, size_t ws_size,
                              hipStream_t stream) {
  (void)n_in; (void)out_size; (void)ws_size;
  const float* x_all  = (const float*)d_in[0];
  const int*   eidx   = (const int*)d_in[1];
  const float* sec1_w = (const float*)d_in[2];  const float* sec1_b = (const float*)d_in[3];
  const float* sec2_w = (const float*)d_in[4];  const float* sec2_b = (const float*)d_in[5];
  const float* gat1_w = (const float*)d_in[6];  const float* gat1_as = (const float*)d_in[7];
  const float* gat1_ad = (const float*)d_in[8]; const float* gat1_b = (const float*)d_in[9];
  const float* gat2_w = (const float*)d_in[10]; const float* gat2_as = (const float*)d_in[11];
  const float* gat2_ad = (const float*)d_in[12]; const float* gat2_b = (const float*)d_in[13];
  const float* gat3_w = (const float*)d_in[14]; const float* gat3_as = (const float*)d_in[15];
  const float* gat3_ad = (const float*)d_in[16]; const float* gat3_b = (const float*)d_in[17];
  const float* wih0 = (const float*)d_in[18];
  const float* bih0 = (const float*)d_in[20]; const float* bhh0 = (const float*)d_in[21];
  const float* wih1 = (const float*)d_in[22];
  const float* bih1 = (const float*)d_in[24]; const float* bhh1 = (const float*)d_in[25];
  const float* wih2 = (const float*)d_in[26];
  const float* bih2 = (const float*)d_in[28]; const float* bhh2 = (const float*)d_in[29];
  const float* time1_w = (const float*)d_in[30]; const float* time1_b = (const float*)d_in[31];
  const float* time2_w = (const float*)d_in[32]; const float* time2_b = (const float*)d_in[33];
  const float* lin1_w = (const float*)d_in[34]; const float* lin1_b = (const float*)d_in[35];
  const float* lin2_w = (const float*)d_in[36]; const float* lin2_b = (const float*)d_in[37];
  const float* lin3_w = (const float*)d_in[38]; const float* lin3_b = (const float*)d_in[39];

  const int n    = in_sizes[0] / N_FEAT;   // 65536
  const int nE   = in_sizes[1] / 2;        // 1048576
  const int nTot = nE + n;                 // edges + self-loops
  const int nodeBlocks = (n + 127) / 128;

  size_t off = 0;
  auto alloc = [&](size_t bytes) -> void* {
    void* p = (char*)d_ws + off;
    off += (bytes + 255) & ~(size_t)255;
    return p;
  };
  float* buf0 = (float*)alloc((size_t)n * 128 * 4);
  float* buf1 = (float*)alloc((size_t)n * 128 * 4);
  float* buf2 = (float*)alloc((size_t)n * 128 * 4);
  float* tout = (float*)alloc((size_t)n * 64 * 4);
  float* es = (float*)alloc((size_t)n * 4);
  float* ed = (float*)alloc((size_t)n * 4);
  float* mm = (float*)alloc((size_t)n * 4);
  float* ss = (float*)alloc((size_t)n * 4);
  bf16_t* sec1b = (bf16_t*)alloc(128 * 64 * 2);
  bf16_t* sec2b = (bf16_t*)alloc(128 * 128 * 2);
  bf16_t* g1b = (bf16_t*)alloc(128 * 128 * 2);
  bf16_t* g2b = (bf16_t*)alloc(128 * 128 * 2);
  bf16_t* g3b = (bf16_t*)alloc(64 * 128 * 2);
  bf16_t* w0b = (bf16_t*)alloc(256 * 32 * 2);
  bf16_t* w1b = (bf16_t*)alloc(256 * 64 * 2);
  bf16_t* w2b = (bf16_t*)alloc(256 * 64 * 2);
  bf16_t* t1b = (bf16_t*)alloc(64 * 64 * 2);
  bf16_t* t2b = (bf16_t*)alloc(64 * 64 * 2);
  bf16_t* l1b = (bf16_t*)alloc(64 * 128 * 2);
  bf16_t* l2b = (bf16_t*)alloc(64 * 64 * 2);
  bf16_t* l3b = (bf16_t*)alloc(16 * 64 * 2);
  float* bs0 = (float*)alloc(256 * 4);
  float* bs1 = (float*)alloc(256 * 4);
  float* bs2 = (float*)alloc(256 * 4);

  auto cvt = [&](const float* s, bf16_t* d, int rows, int cols, int ldk, int rowsPad) {
    int tot = rowsPad * ldk;
    k_cvt<<<(tot + 255) / 256, 256, 0, stream>>>(s, d, rows, cols, ldk, rowsPad);
  };
  cvt(sec1_w, sec1b, 128, 40, 64, 128);
  cvt(sec2_w, sec2b, 128, 128, 128, 128);
  cvt(gat1_w, g1b, 128, 128, 128, 128);
  cvt(gat2_w, g2b, 128, 128, 128, 128);
  cvt(gat3_w, g3b, 64, 128, 128, 64);
  cvt(wih0, w0b, 256, 24, 32, 256);
  cvt(wih1, w1b, 256, 64, 64, 256);
  cvt(wih2, w2b, 256, 64, 64, 256);
  cvt(time1_w, t1b, 64, 64, 64, 64);
  cvt(time2_w, t2b, 64, 64, 64, 64);
  cvt(lin1_w, l1b, 64, 128, 128, 64);
  cvt(lin2_w, l2b, 64, 64, 64, 64);
  cvt(lin3_w, l3b, 4, 64, 64, 16);
  k_bsum<<<1, 256, 0, stream>>>(bih0, bhh0, bs0, 256);
  k_bsum<<<1, 256, 0, stream>>>(bih1, bhh1, bs1, 256);
  k_bsum<<<1, 256, 0, stream>>>(bih2, bhh2, bs2, 256);

  k_sec<<<nodeBlocks, 256, 0, stream>>>(x_all, sec1b, sec1_b, sec2b, sec2_b, buf0);
  k_lstm<<<nodeBlocks, 256, 0, stream>>>(x_all, w0b, bs0, w1b, bs1, w2b, bs2,
                                         t1b, time1_b, t2b, time2_b, tout);

  const int* esrc = eidx;
  const int* edst = eidx + nE;
  const int eb1 = (nTot + 255) / 256;
  const int eb8 = (nTot + 7) / 8;

  // GAT1: in=buf0 (no pre-bias), h=buf1, accum=buf2
  k_gat_h<128><<<nodeBlocks, 256, 0, stream>>>(buf0, nullptr, g1b, gat1_as, gat1_ad,
                                               buf1, es, ed, mm, ss, buf2);
  k_edge_max<<<eb1, 256, 0, stream>>>(esrc, edst, es, ed, mm, nE, nTot);
  k_edge_sum<<<eb1, 256, 0, stream>>>(esrc, edst, es, ed, mm, ss, nE, nTot);
  k_edge_scatter<128><<<eb8, 256, 0, stream>>>(esrc, edst, es, ed, mm, ss, buf1, buf2, nE, nTot);

  // GAT2: in = lrelu(buf2 + gat1_b), h=buf1, accum=buf0
  k_gat_h<128><<<nodeBlocks, 256, 0, stream>>>(buf2, gat1_b, g2b, gat2_as, gat2_ad,
                                               buf1, es, ed, mm, ss, buf0);
  k_edge_max<<<eb1, 256, 0, stream>>>(esrc, edst, es, ed, mm, nE, nTot);
  k_edge_sum<<<eb1, 256, 0, stream>>>(esrc, edst, es, ed, mm, ss, nE, nTot);
  k_edge_scatter<128><<<eb8, 256, 0, stream>>>(esrc, edst, es, ed, mm, ss, buf1, buf0, nE, nTot);

  // GAT3: in = lrelu(buf0 + gat2_b), h=buf1 (N x 64), accum=buf2 (N x 64)
  k_gat_h<64><<<nodeBlocks, 256, 0, stream>>>(buf0, gat2_b, g3b, gat3_as, gat3_ad,
                                              buf1, es, ed, mm, ss, buf2);
  k_edge_max<<<eb1, 256, 0, stream>>>(esrc, edst, es, ed, mm, nE, nTot);
  k_edge_sum<<<eb1, 256, 0, stream>>>(esrc, edst, es, ed, mm, ss, nE, nTot);
  k_edge_scatter<64><<<eb8, 256, 0, stream>>>(esrc, edst, es, ed, mm, ss, buf1, buf2, nE, nTot);

  k_final<<<nodeBlocks, 256, 0, stream>>>(buf2, gat3_b, tout, l1b, lin1_b, l2b, lin2_b,
                                          l3b, lin3_b, (float*)d_out);
}
